// Encoder2_1176821039651
// MI455X (gfx1250) — matrix operand order repaired and checked by reading, not compiled, not run on hardware
//
#include <hip/hip_runtime.h>
#include <hip/hip_bf16.h>
#include <stdint.h>

#define N_NODES 50000
#define N_EDGES 1600000
#define X_DIM   2000
#define HID     128
#define NUM_CLS 16
#define KCH1    63          // ceil(2000/32) k-chunks for GEMM1

typedef __bf16 v16bf __attribute__((ext_vector_type(16)));
typedef float  v8f   __attribute__((ext_vector_type(8)));
typedef int    v4i   __attribute__((ext_vector_type(4)));

union FragB { uint4 u[2]; v16bf v; };

// ---- async global->LDS path (guarded; falls back to sync copy) ----
#ifndef ATHENA_TRY_ASYNC
#define ATHENA_TRY_ASYNC 1
#endif
#if ATHENA_TRY_ASYNC && defined(__AMDGCN__) && defined(__has_builtin)
#if __has_builtin(__builtin_amdgcn_global_load_async_to_lds_b128) && \
    __has_builtin(__builtin_amdgcn_s_wait_asynccnt)
#define HAVE_ASYNC 1
// builtin signature: (global int4* src, local int4* dst, imm int offset, imm int cpol)
#define ASYNC_G(p) ((__attribute__((address_space(1))) v4i*)(p))
#define ASYNC_L(p) ((__attribute__((address_space(3))) v4i*)(p))
#endif
#endif
#ifndef HAVE_ASYNC
#define HAVE_ASYNC 0
#endif

// native f32 -> bf16 (RNE via v_cvt_pk_bf16_f32)
static __device__ __forceinline__ unsigned short f2bfu(float f) {
  __bf16 b = (__bf16)f;
  return __builtin_bit_cast(unsigned short, b);
}
static __device__ __forceinline__ float bfbits2f(uint32_t hbits) {
  union { uint32_t u; float f; } c; c.u = hbits << 16; return c.f;
}

// Load 8 consecutive floats of a row (zero-padded past kmax).
static __device__ __forceinline__ void load8f(const float* __restrict__ row,
                                              int kb, int kmax, float* __restrict__ f) {
  if (kb + 8 <= kmax) {
    float4 a = *(const float4*)(row + kb);
    float4 b = *(const float4*)(row + kb + 4);
    f[0]=a.x; f[1]=a.y; f[2]=a.z; f[3]=a.w;
    f[4]=b.x; f[5]=b.y; f[6]=b.z; f[7]=b.w;
  } else {
#pragma unroll
    for (int i = 0; i < 8; i++) f[i] = (kb + i < kmax) ? row[kb + i] : 0.0f;
  }
}

// Build the 16-bit A fragment for one k-chunk: lane holds row (lane&15),
// e[0..7] = K = k0 + half*8 + i ; e[8..15] = K = k0 + 16 + half*8 + i
static __device__ __forceinline__ FragB make_afrag(const float* __restrict__ row,
                                                   int ch, int half) {
  float f0[8], f1[8];
  const int k0 = ch * 32;
  load8f(row, k0 + half * 8, X_DIM, f0);
  load8f(row, k0 + 16 + half * 8, X_DIM, f1);
  FragB a;
#pragma unroll
  for (int i = 0; i < 8; i++) { a.v[i] = (__bf16)f0[i]; a.v[8 + i] = (__bf16)f1[i]; }
  return a;
}

// ---------------- prep kernels ----------------
__global__ void k_zero(float* __restrict__ p, int n) {
  int i = blockIdx.x * blockDim.x + threadIdx.x;
  if (i < n) p[i] = 0.0f;
}

// W1 [2000,128] -> bf16 B-fragments, layout [ch][nt][lane][16] (zero-padded K>=2000)
__global__ void k_w1frag(const float* __restrict__ W1, unsigned short* __restrict__ frag) {
  int tid = blockIdx.x * blockDim.x + threadIdx.x;
  if (tid >= KCH1 * 8 * 32) return;
  int lane = tid & 31, rest = tid >> 5;
  int nt = rest & 7, ch = rest >> 3;
  int col = nt * 16 + (lane & 15);
  int kb  = ch * 32 + (lane >> 4) * 8;      // B layout: e[0..7] K=half*8+i, e[8..15] K=16+half*8+i
  unsigned short* o = frag + (size_t)tid * 16;
#pragma unroll
  for (int i = 0; i < 16; i++) {
    int k = kb + (i < 8 ? i : i + 8);
    o[i] = f2bfu(k < X_DIM ? W1[(size_t)k * HID + col] : 0.0f);
  }
}

// Wl/Wr [128,128] -> bf16 B-fragments, layout [ch(4)][nt(8)][lane(32)][16]
__global__ void k_wlrfrag(const float* __restrict__ Wl, const float* __restrict__ Wr,
                          unsigned short* __restrict__ fl, unsigned short* __restrict__ fr) {
  int tid = blockIdx.x * blockDim.x + threadIdx.x;
  if (tid >= 2 * 1024) return;
  int sel = tid >> 10, id = tid & 1023;
  int lane = id & 31, rest = id >> 5;
  int nt = rest & 7, ch = rest >> 3;
  int col = nt * 16 + (lane & 15);
  int kb  = ch * 32 + (lane >> 4) * 8;
  const float* W = sel ? Wr : Wl;
  unsigned short* o = (sel ? fr : fl) + (size_t)id * 16;
#pragma unroll
  for (int i = 0; i < 16; i++) o[i] = f2bfu(W[(size_t)(kb + (i < 8 ? i : i + 8)) * HID + col]);
}

// column-normalize Wn [128,16]
__global__ void k_wnhat(const float* __restrict__ Wn, float* __restrict__ wnhat) {
  int c = threadIdx.x;
  if (c >= NUM_CLS) return;
  float ss = 0.0f;
  for (int k = 0; k < HID; k++) { float v = Wn[k * NUM_CLS + c]; ss += v * v; }
  float inv = 1.0f / fmaxf(sqrtf(ss), 1e-12f);
  for (int k = 0; k < HID; k++) wnhat[k * NUM_CLS + c] = Wn[k * NUM_CLS + c] * inv;
}

// ---------------- GEMM1: feat = x@W1 + b1 ; h = relu(feat) (bf16) ----------------
// Double-buffered async B staging; A fragments software-pipelined one chunk ahead.
__global__ __launch_bounds__(256)
void k_gemm1(const float* __restrict__ x, const float* __restrict__ b1,
             const uint4* __restrict__ wb1frag,       // [ch][nt][lane] * 2 uint4
             float* __restrict__ feat, unsigned short* __restrict__ hbf) {
  __shared__ uint4 smemB[2][512];                     // 2 x 8 KB B-fragment buffers
  const int t    = threadIdx.x;
  const int lane = t & 31, w = t >> 5;
  const int half = lane >> 4;
  const int m0   = blockIdx.x * 128;
  const int arow_i = m0 + w * 16 + (lane & 15);
  const float* arow = x + (size_t)min(arow_i, N_NODES - 1) * X_DIM;

  v8f zero = {0.f,0.f,0.f,0.f,0.f,0.f,0.f,0.f};
  v8f acc[8];
#pragma unroll
  for (int nt = 0; nt < 8; nt++) acc[nt] = zero;

#if HAVE_ASYNC
  {  // prologue: stage chunk 0 into buffer 0
    const uint4* s0 = wb1frag;
    __builtin_amdgcn_global_load_async_to_lds_b128(ASYNC_G(s0 + t),
                                                   ASYNC_L(&smemB[0][t]), 0, 0);
    __builtin_amdgcn_global_load_async_to_lds_b128(ASYNC_G(s0 + 256 + t),
                                                   ASYNC_L(&smemB[0][256 + t]), 0, 0);
  }
#endif
  FragB aNext = make_afrag(arow, 0, half);

  for (int ch = 0; ch < KCH1; ch++) {
    const uint4* buf = smemB[ch & 1];
#if HAVE_ASYNC
    __builtin_amdgcn_s_wait_asynccnt(0);   // this wave's share of chunk ch has landed
    __syncthreads();                       // everyone's share landed; prev reads done
    if (ch + 1 < KCH1) {                   // overlap: stage chunk ch+1 during compute
      const uint4* sn = wb1frag + (size_t)(ch + 1) * 512;
      uint4* dn = (uint4*)smemB[(ch + 1) & 1];
      __builtin_amdgcn_global_load_async_to_lds_b128(ASYNC_G(sn + t),
                                                     ASYNC_L(dn + t), 0, 0);
      __builtin_amdgcn_global_load_async_to_lds_b128(ASYNC_G(sn + 256 + t),
                                                     ASYNC_L(dn + 256 + t), 0, 0);
    }
#else
    __syncthreads();
    {
      const uint4* sn = wb1frag + (size_t)ch * 512;
      uint4* dn = (uint4*)smemB[ch & 1];
      dn[t] = sn[t]; dn[256 + t] = sn[256 + t];
    }
    __syncthreads();
#endif

    FragB a = aNext;
    if (ch + 1 < KCH1) {
      aNext = make_afrag(arow, ch + 1, half);
      __builtin_prefetch(arow + (ch + 2) * 32, 0, 1);
    }

    // preload B fragments in groups of 4 -> ds-load clause, then 4 WMMAs back-to-back
#pragma unroll
    for (int g = 0; g < 2; g++) {
      FragB bf[4];
#pragma unroll
      for (int j = 0; j < 4; j++) {
        const int nt = g * 4 + j;
        bf[j].u[0] = buf[nt * 64 + lane * 2];
        bf[j].u[1] = buf[nt * 64 + lane * 2 + 1];
      }
#pragma unroll
      for (int j = 0; j < 4; j++) {
        acc[g * 4 + j] = __builtin_amdgcn_wmma_f32_16x16x32_bf16(
            false, a.v, false, bf[j].v, (short)0, acc[g * 4 + j], false, false);
      }
    }
  }

  // epilogue: C layout — VGPR v: row = v + (lane>>4)*8 ; col = lane&15
  const int colb = lane & 15;
  const int roff = half * 8;
#pragma unroll
  for (int nt = 0; nt < 8; nt++) {
    const int col = nt * 16 + colb;
    const float bb = b1[col];
#pragma unroll
    for (int v = 0; v < 8; v++) {
      int r = m0 + w * 16 + roff + v;
      if (r < N_NODES) {
        float val = acc[nt][v] + bb;
        feat[(size_t)r * HID + col] = val;
        hbf[(size_t)r * HID + col]  = f2bfu(fmaxf(val, 0.0f));
      }
    }
  }
}

// ---------------- scatter: aggr[dst] += h[src] ; cnt[dst] += 1 ----------------
// unsafeAtomicAdd -> native global_atomic_add_f32 (no CAS loop, no return traffic)
__global__ __launch_bounds__(256)
void k_scatter(const long long* __restrict__ ei, const unsigned short* __restrict__ hbf,
               float* __restrict__ aggr, float* __restrict__ cnt) {
  const int lane = threadIdx.x & 31;
  const int wid  = (blockIdx.x * blockDim.x + threadIdx.x) >> 5;
  const int nw   = (gridDim.x * blockDim.x) >> 5;
  for (int e = wid; e < N_EDGES; e += nw) {
    const int src = (int)ei[e];
    const int dst = (int)ei[N_EDGES + e];
    uint2 raw = *(const uint2*)(hbf + (size_t)src * HID + lane * 4);   // 4 bf16
    float* ar = aggr + (size_t)dst * HID + lane * 4;
    unsafeAtomicAdd(ar + 0, bfbits2f(raw.x & 0xFFFFu));
    unsafeAtomicAdd(ar + 1, bfbits2f(raw.x >> 16));
    unsafeAtomicAdd(ar + 2, bfbits2f(raw.y & 0xFFFFu));
    unsafeAtomicAdd(ar + 3, bfbits2f(raw.y >> 16));
    if (lane == 0) unsafeAtomicAdd(cnt + dst, 1.0f);
  }
}

// ---------------- aggrN = bf16(aggr / max(cnt,1)) ----------------
__global__ void k_normalize(const float* __restrict__ aggr, const float* __restrict__ cnt,
                            unsigned short* __restrict__ aggrN) {
  int i = blockIdx.x * blockDim.x + threadIdx.x;
  if (i >= N_NODES * HID) return;
  float inv = 1.0f / fmaxf(cnt[i >> 7], 1.0f);
  aggrN[i] = f2bfu(aggr[i] * inv);
}

// ---------------- h2 = aggrN@Wl + bl + h@Wr ----------------
// Wl/Wr fragments staged once per block into LDS (64 KB of the WGP's 320 KB).
__global__ __launch_bounds__(256)
void k_sage(const unsigned short* __restrict__ aggrN, const unsigned short* __restrict__ hbf,
            const uint4* __restrict__ wlfrag, const uint4* __restrict__ wrfrag,
            const float* __restrict__ bl, float* __restrict__ h2) {
  __shared__ uint4 sB[4096];               // [0..2047] Wl frags, [2048..4095] Wr frags
  const int t = threadIdx.x;
  const int lane = t & 31, w = t >> 5;
  const int half = lane >> 4;
  const int m0 = blockIdx.x * 128;
  const int rr = min(m0 + w * 16 + (lane & 15), N_NODES - 1);
  const uint4* a1row = (const uint4*)(aggrN + (size_t)rr * HID);  // 16 uint4 / row
  const uint4* a2row = (const uint4*)(hbf  + (size_t)rr * HID);

#if HAVE_ASYNC
#pragma unroll
  for (int i = 0; i < 8; i++)
    __builtin_amdgcn_global_load_async_to_lds_b128(ASYNC_G(wlfrag + i * 256 + t),
                                                   ASYNC_L(&sB[i * 256 + t]), 0, 0);
#pragma unroll
  for (int i = 0; i < 8; i++)
    __builtin_amdgcn_global_load_async_to_lds_b128(ASYNC_G(wrfrag + i * 256 + t),
                                                   ASYNC_L(&sB[2048 + i * 256 + t]), 0, 0);
  __builtin_amdgcn_s_wait_asynccnt(0);
#else
#pragma unroll
  for (int i = 0; i < 8; i++) {
    sB[i * 256 + t]        = wlfrag[i * 256 + t];
    sB[2048 + i * 256 + t] = wrfrag[i * 256 + t];
  }
#endif
  __syncthreads();

  v8f zero = {0.f,0.f,0.f,0.f,0.f,0.f,0.f,0.f};
  v8f acc[8];
#pragma unroll
  for (int nt = 0; nt < 8; nt++) acc[nt] = zero;

#pragma unroll
  for (int ch = 0; ch < 4; ch++) {
    const int i0 = ch * 4 + half;          // (ch*32 + half*8)/8 in uint4 units
    FragB a1, a2;
    a1.u[0] = a1row[i0]; a1.u[1] = a1row[i0 + 2];
    a2.u[0] = a2row[i0]; a2.u[1] = a2row[i0 + 2];
#pragma unroll
    for (int nt = 0; nt < 8; nt++) {
      const int idx = ((ch * 8 + nt) * 32 + lane) * 2;
      FragB bL, bR;                        // load both fragments, then both WMMAs
      bL.u[0] = sB[idx];        bL.u[1] = sB[idx + 1];
      bR.u[0] = sB[2048 + idx]; bR.u[1] = sB[2048 + idx + 1];
      acc[nt] = __builtin_amdgcn_wmma_f32_16x16x32_bf16(
          false, a1.v, false, bL.v, (short)0, acc[nt], false, false);
      acc[nt] = __builtin_amdgcn_wmma_f32_16x16x32_bf16(
          false, a2.v, false, bR.v, (short)0, acc[nt], false, false);
    }
  }

  const int colb = lane & 15;
  const int roff = half * 8;
#pragma unroll
  for (int nt = 0; nt < 8; nt++) {
    const int col = nt * 16 + colb;
    const float bb = bl[col];
#pragma unroll
    for (int v = 0; v < 8; v++) {
      int r = m0 + w * 16 + roff + v;
      if (r < N_NODES) h2[(size_t)r * HID + col] = acc[nt][v] + bb;
    }
  }
}

// ---------------- head: out = 10 * l2norm(h2) @ wnhat ----------------
__global__ __launch_bounds__(256)
void k_head(const float* __restrict__ h2, const float* __restrict__ wnhat,
            float* __restrict__ outp) {
  __shared__ float sWn[HID * NUM_CLS];
  for (int i = threadIdx.x; i < HID * NUM_CLS; i += 256) sWn[i] = wnhat[i];
  __syncthreads();
  const int lane = threadIdx.x & 31, w = threadIdx.x >> 5;
  const int row = blockIdx.x * 8 + w;
  if (row >= N_NODES) return;
  const float* hr = h2 + (size_t)row * HID;
  float xv[4]; float ss = 0.0f;
#pragma unroll
  for (int j = 0; j < 4; j++) { xv[j] = hr[lane * 4 + j]; ss += xv[j] * xv[j]; }
#pragma unroll
  for (int m = 16; m >= 1; m >>= 1) ss += __shfl_xor(ss, m, 32);
  const float scale = 10.0f / fmaxf(sqrtf(ss), 1e-12f);
  float o[NUM_CLS];
#pragma unroll
  for (int c = 0; c < NUM_CLS; c++) o[c] = 0.0f;
#pragma unroll
  for (int j = 0; j < 4; j++) {
    float xx = xv[j] * scale;
    const float* wr = sWn + (lane * 4 + j) * NUM_CLS;
#pragma unroll
    for (int c = 0; c < NUM_CLS; c++) o[c] += xx * wr[c];
  }
#pragma unroll
  for (int c = 0; c < NUM_CLS; c++) {
    float v = o[c];
#pragma unroll
    for (int m = 16; m >= 1; m >>= 1) v += __shfl_xor(v, m, 32);
    o[c] = v;
  }
  if (lane == 0) {
#pragma unroll
    for (int c = 0; c < NUM_CLS; c++) outp[(size_t)row * NUM_CLS + c] = o[c];
  }
}

// ---------------- host launcher ----------------
extern "C" void kernel_launch(void* const* d_in, const int* in_sizes, int n_in,
                              void* d_out, int out_size, void* d_ws, size_t ws_size,
                              hipStream_t stream) {
  (void)in_sizes; (void)n_in; (void)out_size; (void)ws_size;
  const float*     x  = (const float*)d_in[0];
  const long long* ei = (const long long*)d_in[1];
  const float*     W1 = (const float*)d_in[2];
  const float*     b1 = (const float*)d_in[3];
  const float*     Wl = (const float*)d_in[4];
  const float*     bl = (const float*)d_in[5];
  const float*     Wr = (const float*)d_in[6];
  const float*     Wn = (const float*)d_in[7];
  float* feat = (float*)d_out;
  float* outp = feat + (size_t)N_NODES * HID;

  char* ws = (char*)d_ws;
  size_t off = 0;
  auto alloc = [&](size_t bytes) {
    off = (off + 255) & ~(size_t)255;
    void* p = ws + off; off += bytes; return p;
  };
  float*          aggr   = (float*)alloc((size_t)N_NODES * HID * 4);  // 25.6 MB
  float*          cnt    = (float*)alloc((size_t)N_NODES * 4);        // contiguous after aggr
  unsigned short* hbf    = (unsigned short*)alloc((size_t)N_NODES * HID * 2);
  unsigned short* aggrN  = (unsigned short*)alloc((size_t)N_NODES * HID * 2);
  float*          h2     = (float*)alloc((size_t)N_NODES * HID * 4);
  unsigned short* wb1    = (unsigned short*)alloc((size_t)KCH1 * 8 * 32 * 16 * 2);
  unsigned short* wlf    = (unsigned short*)alloc((size_t)4 * 8 * 32 * 16 * 2);
  unsigned short* wrf    = (unsigned short*)alloc((size_t)4 * 8 * 32 * 16 * 2);
  float*          wnhat  = (float*)alloc((size_t)HID * NUM_CLS * 4);

  const int nzero = N_NODES * HID + N_NODES;            // aggr + cnt (contiguous)
  k_zero<<<(nzero + 255) / 256, 256, 0, stream>>>(aggr, nzero);
  k_w1frag<<<(KCH1 * 8 * 32 + 255) / 256, 256, 0, stream>>>(W1, wb1);
  k_wlrfrag<<<8, 256, 0, stream>>>(Wl, Wr, wlf, wrf);
  k_wnhat<<<1, 32, 0, stream>>>(Wn, wnhat);

  k_gemm1<<<(N_NODES + 127) / 128, 256, 0, stream>>>(x, b1, (const uint4*)wb1, feat, hbf);
  k_scatter<<<6250, 256, 0, stream>>>(ei, hbf, aggr, cnt);
  k_normalize<<<(N_NODES * HID + 255) / 256, 256, 0, stream>>>(aggr, cnt, aggrN);
  k_sage<<<(N_NODES + 127) / 128, 256, 0, stream>>>(aggrN, hbf, (const uint4*)wlf,
                                                    (const uint4*)wrf, bl, h2);
  k_head<<<(N_NODES + 7) / 8, 256, 0, stream>>>(h2, wnhat, outp);
}
